// MultiHeadAttention_3324304687443
// MI455X (gfx1250) — compile-verified, matrix operand order checked by reading, not run on hardware
//
#include <hip/hip_runtime.h>

// ---------------------------------------------------------------------------
// Types for CDNA5 WMMA (wave32): D = A(16x32 bf16) * B(32x16 bf16) + C(16x16 f32)
// ---------------------------------------------------------------------------
typedef __attribute__((ext_vector_type(16))) __bf16 v16bf;
typedef __attribute__((ext_vector_type(8)))  float  v8f;

union FragB16 {
    uint4 q[2];
    v16bf v;
};

// fp32 -> bf16, round-to-nearest-even
__device__ __forceinline__ unsigned short f2bf(float f) {
    unsigned int u = __float_as_uint(f);
    u += 0x7fffu + ((u >> 16) & 1u);
    return (unsigned short)(u >> 16);
}

// Load a 16-bit A/B fragment from LDS.
// ISA layout: lane holds row (lane&15); k values are two contiguous runs of 8
// starting at kh = (lane>>4)*8 and kh+16.  Caller passes p = &buf[row*stride + kh].
__device__ __forceinline__ v16bf ld_frag(const unsigned short* p) {
    FragB16 f;
    f.q[0] = *(const uint4*)(p);        // k = kh .. kh+7
    f.q[1] = *(const uint4*)(p + 16);   // k = kh+16 .. kh+23
    return f.v;
}

// xor-shuffle across lanes via ds_swizzle (group-of-32: [14:10]=xor, [4:0]=and)
#define SWZ_XOR(v, mask) \
    __int_as_float(__builtin_amdgcn_ds_swizzle(__float_as_int(v), (((mask) << 10) | 0x1f)))

#define WMMA_BF16(a, b, c) \
    __builtin_amdgcn_wmma_f32_16x16x32_bf16(false, (a), false, (b), (short)0, (c), false, false)

// ---------------------------------------------------------------------------
// GEMM:  C[M,N] = X[M,K] @ W[N,K]^T + bias[N]   (torch Linear)
// Block: 256 threads = 8 waves; tile 128x128x32.
// Wave (wm,wn) in 2x4 grid owns 4 m-subtiles x 2 n-subtiles of 16x16.
// ---------------------------------------------------------------------------
#define GBM 128
#define GBN 128
#define GBK 32

__global__ __launch_bounds__(256)
void gemm_bias_bf16_kernel(const float* __restrict__ X, const float* __restrict__ W,
                           const float* __restrict__ bias, float* __restrict__ C,
                           int M, int N, int Kd)
{
    __shared__ unsigned short sA[GBM * GBK];   // [m][k], stride 32
    __shared__ unsigned short sB[GBN * GBK];   // [n][k], stride 32

    const int tid  = threadIdx.x;
    const int lane = tid & 31;
    const int w    = tid >> 5;
    const int wm   = w >> 2;         // 0..1
    const int wn   = w & 3;          // 0..3
    const int l16  = lane & 15;
    const int hg   = lane >> 4;      // half-group
    const int mbase = blockIdx.y * GBM;
    const int nbase = blockIdx.x * GBN;

    v8f acc[4][2] = {};

    for (int k0 = 0; k0 < Kd; k0 += GBK) {
        __syncthreads();
        // stage X and W tiles (convert fp32 -> bf16)
        #pragma unroll
        for (int i = 0; i < 4; i++) {
            int f4  = tid + i * 256;         // 1024 float4 per tile
            int row = f4 >> 3;               // 8 float4 per row of 32
            int c4  = (f4 & 7) << 2;
            float4 xv = *(const float4*)(X + (size_t)(mbase + row) * Kd + k0 + c4);
            uint2 pa;
            pa.x = (unsigned)f2bf(xv.x) | ((unsigned)f2bf(xv.y) << 16);
            pa.y = (unsigned)f2bf(xv.z) | ((unsigned)f2bf(xv.w) << 16);
            *(uint2*)&sA[row * GBK + c4] = pa;
            float4 wv = *(const float4*)(W + (size_t)(nbase + row) * Kd + k0 + c4);
            uint2 pb;
            pb.x = (unsigned)f2bf(wv.x) | ((unsigned)f2bf(wv.y) << 16);
            pb.y = (unsigned)f2bf(wv.z) | ((unsigned)f2bf(wv.w) << 16);
            *(uint2*)&sB[row * GBK + c4] = pb;
        }
        __syncthreads();

        v16bf af[4], bf[2];
        #pragma unroll
        for (int t = 0; t < 4; t++)
            af[t] = ld_frag(&sA[(wm * 64 + t * 16 + l16) * GBK + hg * 8]);
        #pragma unroll
        for (int u = 0; u < 2; u++)
            bf[u] = ld_frag(&sB[(wn * 32 + u * 16 + l16) * GBK + hg * 8]);

        #pragma unroll
        for (int t = 0; t < 4; t++)
            #pragma unroll
            for (int u = 0; u < 2; u++)
                acc[t][u] = WMMA_BF16(af[t], bf[u], acc[t][u]);
    }

    // epilogue: C element (lane,j): row = j + 8*hg, col = l16
    #pragma unroll
    for (int t = 0; t < 4; t++) {
        #pragma unroll
        for (int u = 0; u < 2; u++) {
            int n = nbase + wn * 32 + u * 16 + l16;
            float bv = bias[n];
            #pragma unroll
            for (int j = 0; j < 8; j++) {
                int m = mbase + wm * 64 + t * 16 + j + hg * 8;
                C[(size_t)m * N + n] = acc[t][u][j] + bv;
            }
        }
    }
}

// ---------------------------------------------------------------------------
// Flash attention: one block = one (b,h) and a 128-row Q tile.
// 8 waves; wave w owns score rows w*16..w*16+15 across all 128 kv columns.
// Online softmax in registers (row state replicated across half-group lanes).
// ---------------------------------------------------------------------------
#define AQT 128   // query tile
#define AKT 128   // kv tile
#define AHD 64    // head dim
#define AL  2048
#define AD  1024

__global__ __launch_bounds__(256)
void attn_flash_kernel(const float* __restrict__ Q, const float* __restrict__ Kp,
                       const float* __restrict__ V, float* __restrict__ Octx)
{
    __shared__ unsigned short sQ [AQT * AHD];   // 16 KB  [q][d]  (pre-scaled)
    __shared__ unsigned short sK [AKT * AHD];   // 16 KB  [kv][d]
    __shared__ unsigned short sVt[AHD * AKT];   // 16 KB  [d][kv]
    __shared__ unsigned short sP [AQT * AKT];   // 32 KB  [q][kv] probs (bf16)

    const int tid  = threadIdx.x;
    const int lane = tid & 31;
    const int w    = tid >> 5;
    const int l16  = lane & 15;
    const int hg   = lane >> 4;

    const int bh    = blockIdx.y;
    const int b     = bh >> 4;
    const int h     = bh & 15;
    const int qbase = blockIdx.x * AQT;

    const float* Qg = Q  + ((size_t)(b * AL + qbase)) * AD + h * AHD;
    const float* Kg = Kp + ((size_t)(b * AL)) * AD + h * AHD;
    const float* Vg = V  + ((size_t)(b * AL)) * AD + h * AHD;

    // stage Q tile once, folding in 1/sqrt(64) = 0.125
    #pragma unroll
    for (int i = 0; i < 8; i++) {
        int f4  = tid + i * 256;          // 2048 float4
        int row = f4 >> 4;                // 16 float4 per 64-wide row
        int c4  = (f4 & 15) << 2;
        float4 qv = *(const float4*)(Qg + (size_t)row * AD + c4);
        uint2 pk;
        pk.x = (unsigned)f2bf(qv.x * 0.125f) | ((unsigned)f2bf(qv.y * 0.125f) << 16);
        pk.y = (unsigned)f2bf(qv.z * 0.125f) | ((unsigned)f2bf(qv.w * 0.125f) << 16);
        *(uint2*)&sQ[row * AHD + c4] = pk;
    }

    float mrow[8], lrow[8];
    #pragma unroll
    for (int j = 0; j < 8; j++) { mrow[j] = -3.0e38f; lrow[j] = 0.0f; }
    v8f oacc[4] = {};

    for (int kv0 = 0; kv0 < AL; kv0 += AKT) {
        __syncthreads();   // protect sK/sVt (previous iteration's reads done)
        #pragma unroll
        for (int i = 0; i < 8; i++) {
            int f4  = tid + i * 256;
            int row = f4 >> 4;
            int c4  = (f4 & 15) << 2;
            float4 kv = *(const float4*)(Kg + (size_t)(kv0 + row) * AD + c4);
            uint2 pk;
            pk.x = (unsigned)f2bf(kv.x) | ((unsigned)f2bf(kv.y) << 16);
            pk.y = (unsigned)f2bf(kv.z) | ((unsigned)f2bf(kv.w) << 16);
            *(uint2*)&sK[row * AHD + c4] = pk;
            float4 vv = *(const float4*)(Vg + (size_t)(kv0 + row) * AD + c4);
            sVt[(c4 + 0) * AKT + row] = f2bf(vv.x);
            sVt[(c4 + 1) * AKT + row] = f2bf(vv.y);
            sVt[(c4 + 2) * AKT + row] = f2bf(vv.z);
            sVt[(c4 + 3) * AKT + row] = f2bf(vv.w);
        }
        __syncthreads();

        // ---- scores: S = Qtile @ Ktile^T (rows w*16.., 8 n-subtiles) ----
        v8f s[8] = {};
        #pragma unroll
        for (int ks = 0; ks < 2; ks++) {                 // head dim 64 = 2x32
            v16bf aq = ld_frag(&sQ[(w * 16 + l16) * AHD + ks * 32 + hg * 8]);
            #pragma unroll
            for (int u = 0; u < 8; u++) {
                v16bf bk = ld_frag(&sK[(u * 16 + l16) * AHD + ks * 32 + hg * 8]);
                s[u] = WMMA_BF16(aq, bk, s[u]);
            }
        }

        // ---- online softmax, per accumulator row j (row = w*16 + j + 8*hg) ----
        #pragma unroll
        for (int j = 0; j < 8; j++) {
            float tmax = -3.0e38f;
            #pragma unroll
            for (int u = 0; u < 8; u++) tmax = fmaxf(tmax, s[u][j]);
            tmax = fmaxf(tmax, SWZ_XOR(tmax, 1));
            tmax = fmaxf(tmax, SWZ_XOR(tmax, 2));
            tmax = fmaxf(tmax, SWZ_XOR(tmax, 4));
            tmax = fmaxf(tmax, SWZ_XOR(tmax, 8));

            float newm  = fmaxf(mrow[j], tmax);
            float alpha = __expf(mrow[j] - newm);
            mrow[j] = newm;

            float rsum = 0.0f;
            #pragma unroll
            for (int u = 0; u < 8; u++) {
                float p = __expf(s[u][j] - newm);
                s[u][j] = p;
                rsum += p;
            }
            rsum += SWZ_XOR(rsum, 1);
            rsum += SWZ_XOR(rsum, 2);
            rsum += SWZ_XOR(rsum, 4);
            rsum += SWZ_XOR(rsum, 8);
            lrow[j] = lrow[j] * alpha + rsum;

            #pragma unroll
            for (int v = 0; v < 4; v++) oacc[v][j] *= alpha;

            int rowg = w * 16 + j + 8 * hg;
            #pragma unroll
            for (int u = 0; u < 8; u++)
                sP[rowg * AKT + u * 16 + l16] = f2bf(s[u][j]);
        }
        __syncthreads();   // sP visible (cross-lane within wave)

        // ---- O += P @ Vtile  (kv reduction 128 = 4x32, 4 d-subtiles) ----
        #pragma unroll
        for (int ks = 0; ks < 4; ks++) {
            v16bf ap = ld_frag(&sP[(w * 16 + l16) * AKT + ks * 32 + hg * 8]);
            #pragma unroll
            for (int v = 0; v < 4; v++) {
                v16bf bv = ld_frag(&sVt[(v * 16 + l16) * AKT + ks * 32 + hg * 8]);
                oacc[v] = WMMA_BF16(ap, bv, oacc[v]);
            }
        }
    }

    // epilogue: ctx[b, q, h*64 + d] = O / l   (head merge is implicit)
    float* Og = Octx + ((size_t)(b * AL + qbase)) * AD + h * AHD;
    #pragma unroll
    for (int j = 0; j < 8; j++) {
        float inv = 1.0f / lrow[j];
        int rowl = w * 16 + j + 8 * hg;
        #pragma unroll
        for (int v = 0; v < 4; v++)
            Og[(size_t)rowl * AD + v * 16 + l16] = oacc[v][j] * inv;
    }
}

// ---------------------------------------------------------------------------
extern "C" void kernel_launch(void* const* d_in, const int* in_sizes, int n_in,
                              void* d_out, int out_size, void* d_ws, size_t ws_size,
                              hipStream_t stream) {
    (void)in_sizes; (void)n_in; (void)out_size; (void)ws_size;
    const float* q  = (const float*)d_in[0];
    const float* k  = (const float*)d_in[1];
    const float* v  = (const float*)d_in[2];
    const float* Wq = (const float*)d_in[3];
    const float* bq = (const float*)d_in[4];
    const float* Wk = (const float*)d_in[5];
    const float* bk = (const float*)d_in[6];
    const float* Wv = (const float*)d_in[7];
    const float* bv = (const float*)d_in[8];
    const float* Wo = (const float*)d_in[9];
    const float* bo = (const float*)d_in[10];
    float* out = (float*)d_out;

    const size_t MAT = 4096ull * 1024ull;     // elements per projected matrix
    float* Qp = (float*)d_ws;
    float* Kp = Qp + MAT;
    float* Vp = Kp + MAT;
    float* Cx = Vp + MAT;

    const int M = 4096, N = 1024, Kd = 1024;
    dim3 gblk(256);
    dim3 ggrid(N / GBN, M / GBM);             // (8, 32)

    gemm_bias_bf16_kernel<<<ggrid, gblk, 0, stream>>>(q, Wq, bq, Qp, M, N, Kd);
    gemm_bias_bf16_kernel<<<ggrid, gblk, 0, stream>>>(k, Wk, bk, Kp, M, N, Kd);
    gemm_bias_bf16_kernel<<<ggrid, gblk, 0, stream>>>(v, Wv, bv, Vp, M, N, Kd);

    dim3 agrid(AL / AQT, 2 * 16);             // (16 q-tiles, B*H = 32)
    attn_flash_kernel<<<agrid, gblk, 0, stream>>>(Qp, Kp, Vp, Cx);

    gemm_bias_bf16_kernel<<<ggrid, gblk, 0, stream>>>(Cx, Wo, bo, out, M, N, Kd);
}
